// LatentAttention_70265664962724
// MI455X (gfx1250) — matrix operand order repaired and checked by reading, not compiled, not run on hardware
//
#include <hip/hip_runtime.h>
#include <hip/hip_bf16.h>

// ---------------------------------------------------------------------------
// MLA (latent attention) for MI455X / gfx1250.
// bf16 WMMA for all GEMMs + attention; double-buffered TDM feeds LDS.
// B=2, T=2048, C=2048, H=16, HD=128, RD=64, QLR=1536, KLR=512
// ---------------------------------------------------------------------------

typedef __bf16 bf16;
typedef __attribute__((ext_vector_type(16))) __bf16 v16bf;
typedef __attribute__((ext_vector_type(8)))  __bf16 v8bf;
typedef __attribute__((ext_vector_type(8)))  float  v8f;
typedef __attribute__((ext_vector_type(4)))  unsigned int u32x4;
typedef __attribute__((ext_vector_type(8)))  int i32x8;
typedef __attribute__((ext_vector_type(4)))  int i32x4;

#define WMMA_BF16(a, b, c) \
  __builtin_amdgcn_wmma_f32_16x16x32_bf16(false, (a), false, (b), (short)0, (c), false, false)

#if defined(__AMDGCN__) && __has_builtin(__builtin_amdgcn_tensor_load_to_lds) && \
    __has_builtin(__builtin_amdgcn_s_wait_tensorcnt)
#define HAS_TDM 1
#else
#define HAS_TDM 0
#endif

#define NB     4096      // B*T
#define TT     2048      // T
#define CC     2048      // C
#define HH     16        // heads
#define HD_    128
#define RD_    64
#define QLR_   1536
#define KLR_   512
#define DQK    192       // HD+RD
#define SCALE_ 0.07216878364870322f  // (HD+RD)^-0.5

// ---------------------------------------------------------------------------
// fp32 -> bf16 elementwise (row-major, used for the activation matrix x)
// ---------------------------------------------------------------------------
__global__ __launch_bounds__(256)
void mla_f2bf(const float* __restrict__ in, bf16* __restrict__ out, size_t n) {
  size_t i = (size_t)blockIdx.x * blockDim.x + threadIdx.x;
  if (i < n) out[i] = (bf16)in[i];
}

// ---------------------------------------------------------------------------
// fp32 -> bf16 with transpose: in = W[K][N] row-major, out = Wt[N][K].
// Tiled 32x32 LDS transpose; K,N multiples of 32. 256 threads.
// ---------------------------------------------------------------------------
__global__ __launch_bounds__(256)
void mla_f2bf_t(const float* __restrict__ in, bf16* __restrict__ out, int K, int N) {
  __shared__ float tile[32][33];
  const int k0 = blockIdx.y * 32, n0 = blockIdx.x * 32;
  const int tx = threadIdx.x & 31, ty = threadIdx.x >> 5;   // ty: 0..7
  for (int i = ty; i < 32; i += 8)
    tile[i][tx] = in[(size_t)(k0 + i) * N + n0 + tx];
  __syncthreads();
  for (int i = ty; i < 32; i += 8)
    out[(size_t)(n0 + i) * K + k0 + tx] = (bf16)tile[tx][i];
}

// ---------------------------------------------------------------------------
// GEMM: C[M,N] = A[M,K] * Bt[N,K]^T.  A row-major, Bt = B transposed.
// 128 threads (4 waves). Block tile 128x64, K-step 32.
// Wave w computes rows [w*32, w*32+32) x 64 cols: 8 WMMA accumulators.
// LDS tiles are verbatim 2D copies, streamed by the Tensor Data Mover with
// double buffering: step i issues tile i+1 (buf^1) before waiting
// s_wait_tensorcnt<=2, which (TDM in-order per wave) guarantees the current
// buffer's two descriptors are complete while the next two stay in flight.
// Requires M%128==0, N%64==0, K%32==0 (true for all calls here).
// ---------------------------------------------------------------------------
template <bool OUT_BF16>
__global__ __launch_bounds__(128)
void mla_gemm(const bf16* __restrict__ A, const bf16* __restrict__ Bt,
              void* __restrict__ Cout, int M, int N, int K) {
#if HAS_TDM
  __shared__ bf16 As[2][128 * 32];
  __shared__ bf16 Bs[2][64 * 32];
#else
  __shared__ bf16 As[1][128 * 32];
  __shared__ bf16 Bs[1][64 * 32];
#endif

  const int tid  = threadIdx.x;
  const int lane = tid & 31;
  const int wave = tid >> 5;
  const int half = lane >> 4;
  const int l16  = lane & 15;
  const int bm   = blockIdx.y * 128;
  const int bn   = blockIdx.x * 64;

  v8f acc[2][4];
#pragma unroll
  for (int rr = 0; rr < 2; ++rr)
#pragma unroll
    for (int s = 0; s < 4; ++s)
#pragma unroll
      for (int j = 0; j < 8; ++j) acc[rr][s][j] = 0.0f;

  const int steps = K / 32;

#if HAS_TDM
  // ---- Tensor DMA descriptors (CDNA5 ISA ch.8): 2D tile, data_size=2B ----
  u32x4 gA0, gB0;
  i32x8 gA1, gB1, gz8;
  i32x4 gz;
  gz[0] = gz[1] = gz[2] = gz[3] = 0;
#pragma unroll
  for (int j = 0; j < 8; ++j) gz8[j] = 0;
  // group0: [1:0]=count=1, [63:32]=lds_addr, [120:64]=global_addr, [127:126]=type 2
  gA0[0] = 1u; gA0[1] = 0; gA0[2] = 0; gA0[3] = 0;
  gB0[0] = 1u; gB0[1] = 0; gB0[2] = 0; gB0[3] = 0;
  // group1: data_size=2B, tensor_dim0=K, tensor_dim1=rows, tile=(32 x rowsTile),
  // tensor_dim0_stride=K
  gA1[0] = 0x00010000;                       // data_size=1 (2 bytes)
  gA1[1] = (K & 0xFFFF) << 16;               // tensor_dim0 lo16 -> [63:48]
  gA1[2] = ((unsigned)K >> 16) | ((M & 0xFFFF) << 16);  // dim0 hi16 | dim1 lo16
  gA1[3] = ((unsigned)M >> 16) | (32u << 16);           // dim1 hi16 | tile_dim0=32
  gA1[4] = 128;                              // tile_dim1=128, tile_dim2=0
  gA1[5] = K;                                // tensor_dim0_stride lo32
  gA1[6] = 0; gA1[7] = 0;
  gB1[0] = 0x00010000;
  gB1[1] = (K & 0xFFFF) << 16;
  gB1[2] = ((unsigned)K >> 16) | ((N & 0xFFFF) << 16);
  gB1[3] = ((unsigned)N >> 16) | (32u << 16);
  gB1[4] = 64;                               // tile_dim1=64
  gB1[5] = K;
  gB1[6] = 0; gB1[7] = 0;

  auto tdm_issue = [&](int buf, int k0) {
    unsigned long long ga = (unsigned long long)(size_t)(A + (size_t)bm * K + k0);
    unsigned long long gb = (unsigned long long)(size_t)(Bt + (size_t)bn * K + k0);
    gA0[1] = (unsigned)(size_t)&As[buf][0];
    gA0[2] = (unsigned)ga;
    gA0[3] = ((unsigned)((ga >> 32) & 0x01FFFFFFull)) | 0x80000000u;
    gB0[1] = (unsigned)(size_t)&Bs[buf][0];
    gB0[2] = (unsigned)gb;
    gB0[3] = ((unsigned)((gb >> 32) & 0x01FFFFFFull)) | 0x80000000u;
    __builtin_amdgcn_tensor_load_to_lds(gA0, gA1, gz, gz, gz8, 0);
    __builtin_amdgcn_tensor_load_to_lds(gB0, gB1, gz, gz, gz8, 0);
  };

  if (wave == 0) tdm_issue(0, 0);            // prologue: tile 0 -> buf 0
#endif

  for (int i = 0; i < steps; ++i) {
#if HAS_TDM
    const int buf = i & 1;
    if (wave == 0) {
      if (i + 1 < steps) {
        tdm_issue(buf ^ 1, (i + 1) * 32);    // overlap next tile with compute
        __builtin_amdgcn_s_wait_tensorcnt(2);  // current buf's 2 descriptors done
      } else {
        __builtin_amdgcn_s_wait_tensorcnt(0);
      }
    }
    __syncthreads();                         // publish current buf to all waves
#else
    const int buf = 0;
    const int k0 = i * 32;
    __syncthreads();
#pragma unroll
    for (int u = 0; u < 4; ++u) {
      int c = tid + 128 * u;                 // 0..511
      int row = c >> 2, col8 = (c & 3) << 3;
      *(v8bf*)&As[0][row * 32 + col8] =
          *(const v8bf*)&A[(size_t)(bm + row) * K + k0 + col8];
    }
#pragma unroll
    for (int u = 0; u < 2; ++u) {
      int c = tid + 128 * u;                 // 0..255
      int row = c >> 2, col8 = (c & 3) << 3;
      *(v8bf*)&Bs[0][row * 32 + col8] =
          *(const v8bf*)&Bt[(size_t)(bn + row) * K + k0 + col8];
    }
    __syncthreads();
#endif
    const bf16* Asb = &As[buf][0];
    const bf16* Bsb = &Bs[buf][0];

    // A fragments: lane row = lane%16; elems 0..7 => K=8h..8h+7, 8..15 => 16+8h..
    v16bf afr[2];
#pragma unroll
    for (int rr = 0; rr < 2; ++rr) {
      int row = wave * 32 + rr * 16 + l16;
      v8bf lo = *(const v8bf*)&Asb[row * 32 + 8 * half];
      v8bf hi = *(const v8bf*)&Asb[row * 32 + 16 + 8 * half];
#pragma unroll
      for (int j = 0; j < 8; ++j) { afr[rr][j] = lo[j]; afr[rr][8 + j] = hi[j]; }
    }
#pragma unroll
    for (int s = 0; s < 4; ++s) {
      // B fragment: lane col = lane%16; elems 0..7 => K=8h..8h+7, 8..15 => 16+8h..; Bs rows are n, contiguous k
      int n = s * 16 + l16;
      v8bf lo = *(const v8bf*)&Bsb[n * 32 + 8 * half];
      v8bf hi = *(const v8bf*)&Bsb[n * 32 + 16 + 8 * half];
      v16bf bfr;
#pragma unroll
      for (int j = 0; j < 8; ++j) { bfr[j] = lo[j]; bfr[8 + j] = hi[j]; }
      acc[0][s] = WMMA_BF16(afr[0], bfr, acc[0][s]);
      acc[1][s] = WMMA_BF16(afr[1], bfr, acc[1][s]);
    }
#if HAS_TDM
    __syncthreads();   // all waves done reading buf before step i+1 refills it
#endif
  }

  // epilogue: C layout: vgpr r -> M = r + 8*half, N = lane%16
#pragma unroll
  for (int rr = 0; rr < 2; ++rr) {
    const int rbase = bm + wave * 32 + rr * 16 + 8 * half;
#pragma unroll
    for (int s = 0; s < 4; ++s) {
      int col = bn + s * 16 + l16;
#pragma unroll
      for (int r = 0; r < 8; ++r) {
        size_t idx = (size_t)(rbase + r) * N + col;
        if (OUT_BF16) ((bf16*)Cout)[idx] = (bf16)acc[rr][s][r];
        else          ((float*)Cout)[idx] = acc[rr][s][r];
      }
    }
  }
}

// ---------------------------------------------------------------------------
// RMSNorm over last dim D (f32 in -> bf16 out). One block (256 thr) per row.
// ---------------------------------------------------------------------------
__global__ __launch_bounds__(256)
void mla_rms(const float* __restrict__ in, const float* __restrict__ w,
             bf16* __restrict__ out, int D) {
  const int row = blockIdx.x;
  const float* x = in + (size_t)row * D;
  float ss = 0.f;
  for (int i = threadIdx.x; i < D; i += 256) { float v = x[i]; ss += v * v; }
#pragma unroll
  for (int m = 16; m >= 1; m >>= 1) ss += __shfl_xor(ss, m, 32);
  __shared__ float red[8];
  if ((threadIdx.x & 31) == 0) red[threadIdx.x >> 5] = ss;
  __syncthreads();
  if (threadIdx.x < 32) {
    float v = (threadIdx.x < 8) ? red[threadIdx.x] : 0.f;
#pragma unroll
    for (int m = 4; m >= 1; m >>= 1) v += __shfl_xor(v, m, 32);
    if (threadIdx.x == 0) red[0] = v;
  }
  __syncthreads();
  float inv = rsqrtf(red[0] / (float)D + 1e-6f);
  for (int i = threadIdx.x; i < D; i += 256)
    out[(size_t)row * D + i] = (bf16)(x[i] * inv * w[i]);
}

// ---------------------------------------------------------------------------
// k_pe: RMSNorm over 64 dims + RoPE, one 64-thread block per row.
// ---------------------------------------------------------------------------
__global__ __launch_bounds__(64)
void mla_rms_rope_kpe(const float* __restrict__ in, const float* __restrict__ w,
                      bf16* __restrict__ out) {
  const int row = blockIdx.x;
  const int t   = row & (TT - 1);
  const int i   = threadIdx.x;
  __shared__ float s[64];
  __shared__ float red[2];
  float v  = in[(size_t)row * 64 + i];
  float ss = v * v;
#pragma unroll
  for (int m = 16; m >= 1; m >>= 1) ss += __shfl_xor(ss, m, 32);
  if ((i & 31) == 0) red[i >> 5] = ss;
  s[i] = v;
  __syncthreads();
  float inv = rsqrtf((red[0] + red[1]) / 64.f + 1e-6f);
  int   j    = i & 31;
  float freq = __powf(10000.f, -(float)(2 * j) / 64.f);
  float ang  = (float)t * freq;
  float c = __cosf(ang), sn = __sinf(ang);
  float nv    = v * inv * w[i];
  float other = s[i ^ 32] * inv * w[i ^ 32];
  float r = (i < 32) ? (nv * c - other * sn) : (nv * c + other * sn);
  out[(size_t)row * 64 + i] = (bf16)r;
}

// ---------------------------------------------------------------------------
// Build qcat[b,h,t,192] = [q_nop | rope(q_pe)].  One thread per (n,h).
// ---------------------------------------------------------------------------
__global__ __launch_bounds__(256)
void mla_repack_q(const bf16* __restrict__ qnop, const float* __restrict__ qpe,
                  bf16* __restrict__ qcat) {
  int idx = blockIdx.x * 256 + threadIdx.x;
  if (idx >= NB * HH) return;
  int n = idx >> 4, h = idx & 15;
  int b = n / TT, t = n % TT;
  bf16* dst = qcat + (((size_t)(b * HH + h) * TT) + t) * DQK;
  const bf16* src = qnop + (size_t)n * (HH * HD_) + h * HD_;
  for (int d = 0; d < HD_; ++d) dst[d] = src[d];
  const float* pe = qpe + (size_t)n * (HH * RD_) + h * RD_;
  for (int i = 0; i < RD_; ++i) {
    int   j    = i & 31;
    float freq = __powf(10000.f, -(float)(2 * j) / 64.f);
    float ang  = (float)t * freq;
    float c = __cosf(ang), sn = __sinf(ang);
    float other = pe[i ^ 32];
    float r = (i < 32) ? (pe[i] * c - other * sn) : (pe[i] * c + other * sn);
    dst[HD_ + i] = (bf16)r;
  }
}

// ---------------------------------------------------------------------------
// Build kcat[b,h,t,192] = [k_nop | k_pe(broadcast)] and vT[b,h,128,T].
// ---------------------------------------------------------------------------
__global__ __launch_bounds__(256)
void mla_repack_kv(const bf16* __restrict__ kv, const bf16* __restrict__ kpe,
                   bf16* __restrict__ kcat, bf16* __restrict__ vT) {
  int idx = blockIdx.x * 256 + threadIdx.x;
  if (idx >= NB * HH) return;
  int n = idx >> 4, h = idx & 15;
  int b = n / TT, t = n % TT;
  bf16* kd = kcat + (((size_t)(b * HH + h) * TT) + t) * DQK;
  const bf16* kn = kv + (size_t)n * (HH * 2 * HD_) + h * HD_;
  for (int d = 0; d < HD_; ++d) kd[d] = kn[d];
  const bf16* kp = kpe + (size_t)n * RD_;
  for (int i = 0; i < RD_; ++i) kd[HD_ + i] = kp[i];
  const bf16* vv = kv + (size_t)n * (HH * 2 * HD_) + HH * HD_ + h * HD_;
  bf16* vd = vT + ((size_t)(b * HH + h) * HD_) * TT + t;
  for (int d = 0; d < HD_; ++d) vd[(size_t)d * TT] = vv[d];
}

// ---------------------------------------------------------------------------
// Causal flash attention, 32 keys per iteration, one wave per 16-query tile.
//  S^T(2 tiles) = K * Q^T : 12 WMMA over d=192
//  online softmax: per-query stats via xor-16 shuffles (C layout halves)
//  out^T += V^T * P^T     : 8 dense WMMA (full K=32, no padding)
// ---------------------------------------------------------------------------
__global__ __launch_bounds__(256)
void mla_attn(const bf16* __restrict__ qcat, const bf16* __restrict__ kcat,
              const bf16* __restrict__ vT, bf16* __restrict__ y) {
  const int lane = threadIdx.x & 31;
  const int wid  = threadIdx.x >> 5;
  const int gw   = blockIdx.x * 8 + wid;     // 0 .. B*H*(T/16)-1
  const int qt   = gw & 127;
  const int bh   = gw >> 7;
  const int q0   = qt * 16;
  const int half = lane >> 4, l16 = lane & 15;

  // Q as B-fragments (col = q = lane%16, K rows = d = 8*half+0..7, 16+8*half+0..7)
  const bf16* qrow = qcat + ((size_t)bh * TT + q0 + l16) * DQK;
  v16bf qf[6];
#pragma unroll
  for (int c = 0; c < 6; ++c) {
    v8bf lo = *(const v8bf*)&qrow[c * 32 + 8 * half];
    v8bf hi = *(const v8bf*)&qrow[c * 32 + 16 + 8 * half];
#pragma unroll
    for (int j = 0; j < 8; ++j) { qf[c][j] = lo[j]; qf[c][8 + j] = hi[j]; }
  }

  const bf16* kbase = kcat + (size_t)bh * TT * DQK;
  const bf16* vbase = vT + (size_t)bh * HD_ * TT;

  float m_i = -3.0e38f, l_i = 0.f;
  v8f o[8];
#pragma unroll
  for (int s = 0; s < 8; ++s)
#pragma unroll
    for (int j = 0; j < 8; ++j) o[s][j] = 0.f;

  const int qg  = q0 + l16;
  const int nkt = ((q0 + 15) >> 5) + 1;      // 32-key tiles covering [0, q0+16)

  for (int kt = 0; kt < nkt; ++kt) {
    const int kb = kt * 32;
    const bf16* krow0 = kbase + (size_t)(kb + l16) * DQK;
    const bf16* krow1 = krow0 + (size_t)16 * DQK;
    __builtin_prefetch(krow0 + (size_t)32 * DQK, 0, 1);

    v8f st0, st1;
#pragma unroll
    for (int j = 0; j < 8; ++j) { st0[j] = 0.f; st1[j] = 0.f; }
#pragma unroll
    for (int c = 0; c < 6; ++c) {
      v8bf lo0 = *(const v8bf*)&krow0[c * 32 + 8 * half];
      v8bf hi0 = *(const v8bf*)&krow0[c * 32 + 16 + 8 * half];
      v8bf lo1 = *(const v8bf*)&krow1[c * 32 + 8 * half];
      v8bf hi1 = *(const v8bf*)&krow1[c * 32 + 16 + 8 * half];
      v16bf kf0, kf1;
#pragma unroll
      for (int j = 0; j < 8; ++j) {
        kf0[j] = lo0[j]; kf0[8 + j] = hi0[j];
        kf1[j] = lo1[j]; kf1[8 + j] = hi1[j];
      }
      st0 = WMMA_BF16(kf0, qf[c], st0);      // keys kb+0..15
      st1 = WMMA_BF16(kf1, qf[c], st1);      // keys kb+16..31
    }
    // scale + causal mask, then per-query max
    float tmax = -3.0e38f;
#pragma unroll
    for (int r = 0; r < 8; ++r) {
      int kg0 = kb + r + 8 * half;
      int kg1 = kg0 + 16;
      float s0 = st0[r] * SCALE_;
      float s1 = st1[r] * SCALE_;
      st0[r] = (kg0 <= qg) ? s0 : -3.0e38f;
      st1[r] = (kg1 <= qg) ? s1 : -3.0e38f;
      tmax = fmaxf(tmax, fmaxf(st0[r], st1[r]));
    }
    tmax = fmaxf(tmax, __shfl_xor(tmax, 16, 32));
    float newm  = fmaxf(m_i, tmax);
    float alpha = __expf(m_i - newm);
    float psum = 0.f;
#pragma unroll
    for (int r = 0; r < 8; ++r) {
      st0[r] = __expf(st0[r] - newm);
      st1[r] = __expf(st1[r] - newm);
      psum += st0[r] + st1[r];
    }
    psum += __shfl_xor(psum, 16, 32);
    l_i = l_i * alpha + psum;
    m_i = newm;
#pragma unroll
    for (int s = 0; s < 8; ++s)
#pragma unroll
      for (int j = 0; j < 8; ++j) o[s][j] *= alpha;

    // P^T B-fragment: lanes 0-15 hold keys kb+0..15, lanes 16-31 keys kb+16..31.
    v16bf pb;
#pragma unroll
    for (int r = 0; r < 8; ++r) {
      float o0 = __shfl_xor(st0[r], 16, 32);
      float o1 = __shfl_xor(st1[r], 16, 32);
      pb[r]     = (half == 0) ? (bf16)st0[r] : (bf16)o1;
      pb[8 + r] = (half == 0) ? (bf16)o0    : (bf16)st1[r];
    }
    // out^T += V^T * P^T  (A rows = dv, K = 32 keys, fully dense)
#pragma unroll
    for (int s = 0; s < 8; ++s) {
      const bf16* vrow = vbase + (size_t)(s * 16 + l16) * TT;
      v8bf lo = *(const v8bf*)&vrow[kb + 16 * half];
      v8bf hi = *(const v8bf*)&vrow[kb + 16 * half + 8];
      v16bf af;
#pragma unroll
      for (int j = 0; j < 8; ++j) { af[j] = lo[j]; af[8 + j] = hi[j]; }
      o[s] = WMMA_BF16(af, pb, o[s]);
    }
  }

  const float invl = 1.f / l_i;
  const int b = bh >> 4, h = bh & 15;
  const int t = q0 + l16;
#pragma unroll
  for (int s = 0; s < 8; ++s)
#pragma unroll
    for (int r = 0; r < 8; ++r) {
      int dv = s * 16 + r + 8 * half;
      y[((size_t)b * TT + t) * (HH * HD_) + h * HD_ + dv] = (bf16)(o[s][r] * invl);
    }
}

// ---------------------------------------------------------------------------
// Host-side orchestration
// ---------------------------------------------------------------------------
extern "C" void kernel_launch(void* const* d_in, const int* in_sizes, int n_in,
                              void* d_out, int out_size, void* d_ws, size_t ws_size,
                              hipStream_t stream) {
  const float* x    = (const float*)d_in[0];
  const float* Wqd  = (const float*)d_in[1];
  const float* qnw  = (const float*)d_in[2];
  const float* Wqu  = (const float*)d_in[3];
  const float* Wqp  = (const float*)d_in[4];
  const float* Wkvd = (const float*)d_in[5];
  const float* kvnw = (const float*)d_in[6];
  const float* Wkvu = (const float*)d_in[7];
  const float* Wkp  = (const float*)d_in[8];
  const float* kpnw = (const float*)d_in[9];
  const float* Wo   = (const float*)d_in[10];
  float* out = (float*)d_out;

  char*  ws  = (char*)d_ws;
  size_t off = 0;
  auto alloc = [&](size_t bytes) -> void* {
    void* p = ws + off;
    off += (bytes + 255) & ~(size_t)255;
    return p;
  };

  bf16* xb    = (bf16*)alloc((size_t)NB * CC * 2);
  bf16* wqdT  = (bf16*)alloc((size_t)CC * QLR_ * 2);            // [QLR][C]
  bf16* wquT  = (bf16*)alloc((size_t)QLR_ * HH * HD_ * 2);      // [H*HD][QLR]
  bf16* wqpT  = (bf16*)alloc((size_t)QLR_ * HH * RD_ * 2);      // [H*RD][QLR]
  bf16* wkvdT = (bf16*)alloc((size_t)CC * KLR_ * 2);            // [KLR][C]
  bf16* wkvuT = (bf16*)alloc((size_t)KLR_ * HH * 2 * HD_ * 2);  // [H*2*HD][KLR]
  bf16* wkpT  = (bf16*)alloc((size_t)CC * RD_ * 2);             // [RD][C]
  bf16* woT   = (bf16*)alloc((size_t)CC * CC * 2);              // [C][H*HD]
  float* cQf  = (float*)alloc((size_t)NB * QLR_ * 4);
  bf16*  cQb  = (bf16*)alloc((size_t)NB * QLR_ * 2);
  bf16*  qnop = (bf16*)alloc((size_t)NB * HH * HD_ * 2);
  float* qpef = (float*)alloc((size_t)NB * HH * RD_ * 4);
  float* cKVf = (float*)alloc((size_t)NB * KLR_ * 4);
  bf16*  cKVb = (bf16*)alloc((size_t)NB * KLR_ * 2);
  bf16*  kvb  = (bf16*)alloc((size_t)NB * HH * 2 * HD_ * 2);
  float* kpef = (float*)alloc((size_t)NB * RD_ * 4);
  bf16*  kpeb = (bf16*)alloc((size_t)NB * RD_ * 2);
  bf16*  qcat = (bf16*)alloc((size_t)2 * HH * TT * DQK * 2);
  bf16*  kcat = (bf16*)alloc((size_t)2 * HH * TT * DQK * 2);
  bf16*  vTb  = (bf16*)alloc((size_t)2 * HH * HD_ * TT * 2);
  bf16*  yb   = (bf16*)alloc((size_t)NB * HH * HD_ * 2);

  // x: plain convert (A operand, row-major)
  mla_f2bf<<<dim3((unsigned)(((size_t)NB * CC + 255) / 256)), dim3(256), 0, stream>>>(
      x, xb, (size_t)NB * CC);
  // weights: convert + transpose (B operands become [N][K])
  auto convT = [&](const float* src, bf16* dst, int K, int N) {
    mla_f2bf_t<<<dim3(N / 32, K / 32), dim3(256), 0, stream>>>(src, dst, K, N);
  };
  convT(Wqd,  wqdT,  CC,   QLR_);
  convT(Wqu,  wquT,  QLR_, HH * HD_);
  convT(Wqp,  wqpT,  QLR_, HH * RD_);
  convT(Wkvd, wkvdT, CC,   KLR_);
  convT(Wkvu, wkvuT, KLR_, HH * 2 * HD_);
  convT(Wkp,  wkpT,  CC,   RD_);
  convT(Wo,   woT,   CC,   CC);

  auto gemmF = [&](const bf16* A, const bf16* Bt, float* C, int M, int N, int K) {
    mla_gemm<false><<<dim3(N / 64, M / 128), dim3(128), 0, stream>>>(A, Bt, (void*)C, M, N, K);
  };
  auto gemmB = [&](const bf16* A, const bf16* Bt, bf16* C, int M, int N, int K) {
    mla_gemm<true><<<dim3(N / 64, M / 128), dim3(128), 0, stream>>>(A, Bt, (void*)C, M, N, K);
  };

  // Q path
  gemmF(xb, wqdT, cQf, NB, QLR_, CC);
  mla_rms<<<dim3(NB), dim3(256), 0, stream>>>(cQf, qnw, cQb, QLR_);
  gemmB(cQb, wquT, qnop, NB, HH * HD_, QLR_);
  gemmF(cQb, wqpT, qpef, NB, HH * RD_, QLR_);
  // KV path
  gemmF(xb, wkvdT, cKVf, NB, KLR_, CC);
  mla_rms<<<dim3(NB), dim3(256), 0, stream>>>(cKVf, kvnw, cKVb, KLR_);
  gemmB(cKVb, wkvuT, kvb, NB, HH * 2 * HD_, KLR_);
  // decoupled RoPE key
  gemmF(xb, wkpT, kpef, NB, RD_, CC);
  mla_rms_rope_kpe<<<dim3(NB), dim3(64), 0, stream>>>(kpef, kpnw, kpeb);
  // repack into per-(b,h) attention layouts
  mla_repack_q<<<dim3(NB * HH / 256), dim3(256), 0, stream>>>(qnop, qpef, qcat);
  mla_repack_kv<<<dim3(NB * HH / 256), dim3(256), 0, stream>>>(kvb, kpeb, kcat, vTb);
  // causal flash attention: B*H*(T/16) waves, 8 waves/block
  mla_attn<<<dim3(2 * HH * (TT / 16) / 8), dim3(256), 0, stream>>>(qcat, kcat, vTb, yb);
  // output projection (f32 out)
  gemmF(yb, woT, out, NB, CC, CC);
}
